// FlexAttnScoreModBlockMaskModel_63797444215440
// MI455X (gfx1250) — matrix operand order repaired and checked by reading, not compiled, not run on hardware
//
#include <hip/hip_runtime.h>
#include <hip/hip_bf16.h>

// ---------------------------------------------------------------------------
// Shapes (fixed by the reference): B=2, S=2048, D=1024, H=16, hd=64, SCALE=.25
// ---------------------------------------------------------------------------
#define BB   2
#define SEQ  2048
#define DM   1024
#define NH   16
#define HD   64
#define MROWS (BB * SEQ)          // 4096

typedef __attribute__((ext_vector_type(16))) __bf16         v16bf;
typedef __attribute__((ext_vector_type(8)))  float          v8f;
typedef __attribute__((ext_vector_type(8)))  unsigned short v8u;
typedef __attribute__((ext_vector_type(4)))  unsigned short v4u;
typedef __attribute__((ext_vector_type(4)))  unsigned int   u32x4;
typedef __attribute__((ext_vector_type(8)))  int            i32x8;
typedef __attribute__((ext_vector_type(4)))  int            i32x4;

__device__ __forceinline__ unsigned short f2bf(float f) {
    unsigned int u = __float_as_uint(f);
    u += 0x7fffu + ((u >> 16) & 1u);          // round-to-nearest-even
    return (unsigned short)(u >> 16);
}

// pack two f32 -> packed bf16x2 (low = a, high = b)
__device__ __forceinline__ unsigned pack_bf16(float a, float b) {
#if __has_builtin(__builtin_amdgcn_cvt_pk_bf16_f32)
    auto p = __builtin_amdgcn_cvt_pk_bf16_f32(a, b);
    return __builtin_bit_cast(unsigned, p);
#else
    return (unsigned)f2bf(a) | ((unsigned)f2bf(b) << 16);
#endif
}

struct V8x2 { v8u lo, hi; };
__device__ __forceinline__ v16bf frag_from(v8u lo, v8u hi) {
    V8x2 t{lo, hi};
    return __builtin_bit_cast(v16bf, t);
}

// A-matrix fragment, 16(M) x 32(K) bf16, row-major source, leading dim ld (elems).
// ISA layout: lanes 0-15 -> M=lane, K={0..7,16..23}; lanes 16-31 -> K={8..15,24..31}.
__device__ __forceinline__ v16bf load_frag_a(const unsigned short* base, int ld, int lane) {
    int m  = lane & 15;
    int k0 = (lane >> 4) << 3;                 // 0 or 8
    const unsigned short* p = base + (size_t)m * ld + k0;
    v8u lo = *(const v8u*)p;                   // K = k0 .. k0+7
    v8u hi = *(const v8u*)(p + 16);            // K = k0+16 .. k0+23
    return frag_from(lo, hi);
}

// B-matrix fragment, 32(K) x 16(N) bf16, memory holds rows n over K (NT pattern).
// ISA layout: lane -> column n=lane&15, K={0..7,16..23} / {8..15,24..31}.
__device__ __forceinline__ v16bf load_frag_b(const unsigned short* base, int ld, int lane) {
    int n    = lane & 15;
    int koff = (lane >> 4) << 3;               // 0 or 8
    const unsigned short* p = base + (size_t)n * ld + koff;
    v8u lo = *(const v8u*)p;
    v8u hi = *(const v8u*)(p + 16);
    return frag_from(lo, hi);
}

__device__ __forceinline__ v8f wmma_bf16(v16bf a, v16bf b, v8f c) {
    return __builtin_amdgcn_wmma_f32_16x16x32_bf16(
        false, a, false, b, (short)0, c, false, false);
}

__device__ __forceinline__ unsigned rfl(unsigned v) {
    return (unsigned)__builtin_amdgcn_readfirstlane((int)v);
}

// ---------------------------------------------------------------------------
// Tensor Data Mover: 2-D tile load, global (row-major, row_stride elems of 2B)
// -> LDS, optional LDS row padding (CDNA5 ISA).
// ---------------------------------------------------------------------------
__device__ __forceinline__ void tdm_load_2d(unsigned lds_addr, const void* gptr,
                                            unsigned tile_d0, unsigned tile_d1,
                                            unsigned row_stride,
                                            unsigned pad_iv, unsigned pad_am,
                                            unsigned pad_en) {
    unsigned long long ga = (unsigned long long)(size_t)gptr;
    unsigned ga_lo = rfl((unsigned)ga);
    unsigned ga_hi = rfl((unsigned)(ga >> 32));
    unsigned lds   = rfl(lds_addr);

    u32x4 g0;
    g0[0] = 1u;                                   // count=1 (valid), user mode
    g0[1] = lds;                                  // lds_addr
    g0[2] = ga_lo;                                // global_addr[31:0]
    g0[3] = (ga_hi & 0x01ffffffu) | (2u << 30);   // global_addr[56:32] | type=2

    i32x8 g1;
    g1[0] = (int)((1u << 16) | (pad_en << 20) |
                  (pad_iv << 22) | (pad_am << 25));     // data_size=2B, pad ctl
    g1[1] = (int)(tile_d0 << 16);                 // tensor_dim0[15:0]
    g1[2] = (int)(tile_d1 << 16);                 // tensor_dim0 hi | tensor_dim1 lo
    g1[3] = (int)(tile_d0 << 16);                 // tensor_dim1 hi | tile_dim0
    g1[4] = (int)tile_d1;                         // tile_dim1 | tile_dim2=0
    g1[5] = (int)row_stride;                      // tensor_dim0_stride[31:0]
    g1[6] = 0;
    g1[7] = 0;

    i32x4 z4 = {0, 0, 0, 0};
#if defined(__clang_major__) && (__clang_major__ >= 23)
    i32x8 z8 = {0, 0, 0, 0, 0, 0, 0, 0};
    __builtin_amdgcn_tensor_load_to_lds(g0, g1, z4, z4, z8, 0);
#else
    __builtin_amdgcn_tensor_load_to_lds(g0, g1, z4, z4, 0);
#endif
}

__device__ __forceinline__ void tdm_wait() {
    __builtin_amdgcn_s_wait_tensorcnt(0);
    __asm__ volatile("" ::: "memory");            // keep LDS reads below the wait
}

// ---------------------------------------------------------------------------
// fp32 -> bf16 conversion
// ---------------------------------------------------------------------------
__global__ void cvt_bf16_kernel(const float* __restrict__ src,
                                unsigned short* __restrict__ dst, int n) {
    int i = blockIdx.x * blockDim.x + threadIdx.x;
    int stride = gridDim.x * blockDim.x;
    for (; i < n; i += stride) dst[i] = f2bf(src[i]);
}

// ---------------------------------------------------------------------------
// NT GEMM:  C[m,n] = sum_k A[m,k] * B[n,k]   (A: MxK bf16, B: NxK bf16)
// Workgroup = 8 waves -> 128(M) x 128(N) tile. TDM double-buffers 128x32
// A/B tiles into LDS (rows padded 64B->80B: conflict-free, 16B-aligned).
// Each wave: 32(M) x 64(N) -> 8 WMMAs per K=32 step from LDS fragments.
// ---------------------------------------------------------------------------
#define GEMM_LD 40                                 // padded LDS row pitch (ushorts)

template <int OUTF32>
__global__ __launch_bounds__(256) void gemm_nt_kernel(
        const unsigned short* __restrict__ A,
        const unsigned short* __restrict__ B,
        float* __restrict__ Cf,
        unsigned short* __restrict__ Cb,
        int M, int N, int K) {
    __shared__ __align__(16) unsigned short As[2][128 * GEMM_LD];
    __shared__ __align__(16) unsigned short Bs[2][128 * GEMM_LD];

    int lane  = threadIdx.x & 31;
    int wslot = threadIdx.x >> 5;
    int wm    = wslot >> 1;                        // 0..3 : M sub-tile
    int wn    = wslot & 1;                         // 0..1 : N sub-tile

    int ntb = N >> 7;                              // N / 128
    int mb  = (int)blockIdx.x / ntb;
    int nb  = (int)blockIdx.x % ntb;
    if (mb * 128 >= M) return;

    const unsigned short* Abase = A + (size_t)(mb * 128) * K;
    const unsigned short* Bbase = B + (size_t)(nb * 128) * K;

    if (wslot == 0) {
        tdm_load_2d((unsigned)(size_t)&As[0][0], Abase, 32, 128, K, 3, 3, 1);
        tdm_load_2d((unsigned)(size_t)&Bs[0][0], Bbase, 32, 128, K, 3, 3, 1);
        tdm_wait();
    }
    __syncthreads();

    v8f acc[2][4] = {};
    for (int k = 0; k < K; k += 32) {
        int cur = (k >> 5) & 1;
        bool more = (k + 32) < K;
        if (wslot == 0 && more) {
            tdm_load_2d((unsigned)(size_t)&As[1 - cur][0], Abase + k + 32, 32, 128, K, 3, 3, 1);
            tdm_load_2d((unsigned)(size_t)&Bs[1 - cur][0], Bbase + k + 32, 32, 128, K, 3, 3, 1);
        }

        const unsigned short* At = &As[cur][0] + (size_t)(wm * 32) * GEMM_LD;
        const unsigned short* Bt = &Bs[cur][0] + (size_t)(wn * 64) * GEMM_LD;

        v16bf a0 = load_frag_a(At,                GEMM_LD, lane);
        v16bf a1 = load_frag_a(At + 16 * GEMM_LD, GEMM_LD, lane);
        v16bf bf[4];
#pragma unroll
        for (int t = 0; t < 4; ++t)
            bf[t] = load_frag_b(Bt + (size_t)(t * 16) * GEMM_LD, GEMM_LD, lane);
#pragma unroll
        for (int t = 0; t < 4; ++t) {
            acc[0][t] = wmma_bf16(a0, bf[t], acc[0][t]);
            acc[1][t] = wmma_bf16(a1, bf[t], acc[1][t]);
        }

        if (wslot == 0 && more) tdm_wait();
        __syncthreads();
    }

    int ncol = lane & 15;
    int rh   = (lane >> 4) << 3;                   // 0 or 8
#pragma unroll
    for (int i = 0; i < 2; ++i)
#pragma unroll
        for (int t = 0; t < 4; ++t)
#pragma unroll
            for (int r = 0; r < 8; ++r) {
                int row = mb * 128 + wm * 32 + i * 16 + rh + r;
                int col = nb * 128 + wn * 64 + t * 16 + ncol;
                if (OUTF32) Cf[(size_t)row * N + col] = acc[i][t][r];
                else        Cb[(size_t)row * N + col] = f2bf(acc[i][t][r]);
            }
}

// ---------------------------------------------------------------------------
// V transpose: Vs row-major [B*S, D] (bf16) -> Vt head-major [B, H, hd, S].
// LDS-tiled 64x64 so both global sides are 8B-vector coalesced.
// ---------------------------------------------------------------------------
__global__ __launch_bounds__(256) void transpose_v_kernel(
        const unsigned short* __restrict__ Vs,
        unsigned short* __restrict__ Vt) {
    __shared__ __align__(16) unsigned short T[64][68];   // 136B pitch: 8B-aligned

    int bh = (int)blockIdx.y;                    // b*NH + h   (0..31)
    int st = (int)blockIdx.x;                    // s-tile of 64 (0..31)
    int b  = bh >> 4;
    int h  = bh & 15;

    int tid = (int)threadIdx.x;
    int c4  = (tid & 15) << 2;                   // 0..60 step 4 (elems)
    int r0  = tid >> 4;                          // 0..15

    const unsigned short* src = Vs + (size_t)(b * SEQ + st * 64) * DM + h * HD;
#pragma unroll
    for (int rr = 0; rr < 64; rr += 16) {
        int r = r0 + rr;
        *(v4u*)&T[r][c4] = *(const v4u*)(src + (size_t)r * DM + c4);
    }
    __syncthreads();

    unsigned short* dst = Vt + (size_t)bh * HD * SEQ + st * 64;
#pragma unroll
    for (int dd = 0; dd < 64; dd += 16) {
        int d = r0 + dd;
        v4u o;
#pragma unroll
        for (int j = 0; j < 4; ++j) o[j] = T[c4 + j][d];
        *(v4u*)(dst + (size_t)d * SEQ + c4) = o;
    }
}

// ---------------------------------------------------------------------------
// Flash attention, transposed-score formulation. One wave per (b,h,16 queries).
// S^T = K·Q^T  (K = A-matrix, Q = B-matrix): each lane owns ONE query column
// and 16 of the 32 keys -> softmax reductions are in-lane VALU trees plus a
// single shfl_xor(16); softmax state m,l are per-lane scalars.
// P^T packs to LDS as two 16B ds_store_b128 per iteration, re-read in
// A-layout for 4 PV WMMAs against contiguous V^T fragments.
// scores = 0.25*qk + (k - q), causal (keep when k <= q).
// ---------------------------------------------------------------------------
#define P_LD 40                                    // padded P row pitch (ushorts)

__global__ __launch_bounds__(256) void attn_kernel(
        const unsigned short* __restrict__ Qg,
        const unsigned short* __restrict__ Kg,
        const unsigned short* __restrict__ Vt,   // [B,H,hd,S]
        unsigned short* __restrict__ Og) {
    __shared__ __align__(16) unsigned short Pl[8][16 * P_LD];    // P staging

    int lane  = threadIdx.x & 31;
    int wslot = threadIdx.x >> 5;
    int wid   = (int)((blockIdx.x * blockDim.x + threadIdx.x) >> 5);
    if (wid >= BB * NH * (SEQ / 16)) return;

    int qb = wid & (SEQ / 16 - 1);               // 0..127
    int h  = (wid >> 7) & (NH - 1);
    int b  = wid >> 11;

    int ncol  = lane & 15;
    int nhalf = lane >> 4;                       // 0/1 -> C-layout row half

    const unsigned short* Qp = Qg + ((size_t)(b * SEQ + qb * 16)) * DM + h * HD;
    // Q as B-matrix (hd x query): lane -> query column ncol
    v16bf qfb0 = load_frag_b(Qp,      DM, lane); // hd 0..31
    v16bf qfb1 = load_frag_b(Qp + 32, DM, lane); // hd 32..63

    const unsigned short* Vtb = Vt + (size_t)(b * NH + h) * HD * SEQ;
    const unsigned short* Kbb = Kg + (size_t)(b * SEQ) * DM + h * HD;

    v8f o[4] = {};
    float mq = -3.0e38f, lq = 0.0f;              // softmax state for query ncol
    float qif = (float)(qb * 16 + ncol);
    float rfv[8];
#pragma unroll
    for (int r = 0; r < 8; ++r) rfv[r] = (float)(r + 8 * nhalf);

    unsigned short* Pw = Pl[wslot];
    int nkb = (qb * 16 + 15) / 32 + 1;           // key blocks of 32

    for (int kb = 0; kb < nkb; ++kb) {
        int key0 = kb * 32;

        // ---- S^T: two 16(key) x 16(query) tiles ---------------------------
        v8f s[2];
#pragma unroll
        for (int h2 = 0; h2 < 2; ++h2) {
            const unsigned short* Kp = Kbb + (size_t)(key0 + h2 * 16) * DM;
            v16bf ka0 = load_frag_a(Kp,      DM, lane);
            v16bf ka1 = load_frag_a(Kp + 32, DM, lane);
            v8f si = {};
            si = wmma_bf16(ka0, qfb0, si);
            si = wmma_bf16(ka1, qfb1, si);
            s[h2] = si;
        }

        // ---- bias + causal mask: element key = key0 + h2*16 + rfv[r] ------
#pragma unroll
        for (int h2 = 0; h2 < 2; ++h2) {
            float fb = (float)(key0 + h2 * 16) - qif;
#pragma unroll
            for (int r = 0; r < 8; ++r) {
                float bb = fb + rfv[r];          // = k - q; keep when <= 0
                float v  = fmaf(s[h2][r], 0.25f, bb);
                s[h2][r] = (bb <= 0.0f) ? v : -3.0e38f;
            }
        }

        // ---- in-lane max over 16 keys, combine opposite half --------------
        float mx = s[0][0];
#pragma unroll
        for (int h2 = 0; h2 < 2; ++h2)
#pragma unroll
            for (int r = 0; r < 8; ++r)
                if (h2 | r) mx = fmaxf(mx, s[h2][r]);
        mx = fmaxf(mx, __shfl_xor(mx, 16, 32));
        float mnew = fmaxf(mq, mx);
        float cfq  = __expf(mq - mnew);
        mq = mnew;

        // ---- exponentiate + in-lane sum + combine -------------------------
        float rs = 0.0f;
#pragma unroll
        for (int h2 = 0; h2 < 2; ++h2)
#pragma unroll
            for (int r = 0; r < 8; ++r) {
                float sv = s[h2][r];
                float p  = (sv > -1.0e37f) ? __expf(sv - mnew) : 0.0f;
                s[h2][r] = p;
                rs += p;
            }
        rs += __shfl_xor(rs, 16, 32);
        lq = lq * cfq + rs;

        // ---- rescale O: broadcast cf of each query row to its owners ------
        float cfr[8];
#pragma unroll
        for (int r = 0; r < 8; ++r)
            cfr[r] = __shfl(cfq, r + 8 * nhalf, 32);
#pragma unroll
        for (int t = 0; t < 4; ++t)
#pragma unroll
            for (int r = 0; r < 8; ++r) o[t][r] *= cfr[r];

        // ---- P^T -> LDS [query][key], packed 16B stores -------------------
        // lane writes keys h2*16 + 8*nhalf + (0..7) of query-row ncol
#pragma unroll
        for (int h2 = 0; h2 < 2; ++h2) {
            union { v8u u16; unsigned u32[4]; } pu;
#pragma unroll
            for (int j = 0; j < 4; ++j)
                pu.u32[j] = pack_bf16(s[h2][2 * j], s[h2][2 * j + 1]);
            *(v8u*)(Pw + ncol * P_LD + h2 * 16 + nhalf * 8) = pu.u16;
        }

        v16bf pf = load_frag_a(Pw, P_LD, lane);  // intra-wave, DScnt-ordered

        // ---- P * V : contiguous fragments from V^T ------------------------
#pragma unroll
        for (int t = 0; t < 4; ++t) {
            v16bf vf = load_frag_b(Vtb + (size_t)(t * 16) * SEQ + key0, SEQ, lane);
            o[t] = wmma_bf16(pf, vf, o[t]);
        }
    }

    // ---- epilogue: normalize (broadcast 1/l per row), store bf16 ----------
    float invl = 1.0f / lq;
    float ir[8];
#pragma unroll
    for (int r = 0; r < 8; ++r)
        ir[r] = __shfl(invl, r + 8 * nhalf, 32);
#pragma unroll
    for (int t = 0; t < 4; ++t)
#pragma unroll
        for (int r = 0; r < 8; ++r) {
            int row = qb * 16 + nhalf * 8 + r;
            int col = h * HD + t * 16 + ncol;
            Og[(size_t)(b * SEQ + row) * DM + col] = f2bf(o[t][r] * ir[r]);
        }
}

// ---------------------------------------------------------------------------
// Host-side orchestration
// ---------------------------------------------------------------------------
extern "C" void kernel_launch(void* const* d_in, const int* in_sizes, int n_in,
                              void* d_out, int out_size, void* d_ws, size_t ws_size,
                              hipStream_t stream) {
    (void)in_sizes; (void)n_in; (void)out_size; (void)ws_size;

    const float* x  = (const float*)d_in[0];
    const float* wq = (const float*)d_in[1];
    const float* wk = (const float*)d_in[2];
    const float* wv = (const float*)d_in[3];
    const float* wo = (const float*)d_in[4];
    float* out = (float*)d_out;

    // bf16 workspace layout (ushort elements), total 24M elems = 48 MB
    unsigned short* xb  = (unsigned short*)d_ws;
    unsigned short* wqb = xb  + (size_t)MROWS * DM;
    unsigned short* wkb = wqb + (size_t)DM * DM;
    unsigned short* wvb = wkb + (size_t)DM * DM;
    unsigned short* wob = wvb + (size_t)DM * DM;
    unsigned short* Qs  = wob + (size_t)DM * DM;
    unsigned short* Ks  = Qs  + (size_t)MROWS * DM;
    unsigned short* Vs  = Ks  + (size_t)MROWS * DM;
    unsigned short* Os  = Vs  + (size_t)MROWS * DM;
    unsigned short* Vt  = xb;   // alias: xb is dead after the projection GEMMs

    // 1) convert inputs to bf16
    cvt_bf16_kernel<<<2048, 256, 0, stream>>>(x,  xb,  MROWS * DM);
    cvt_bf16_kernel<<<1024, 256, 0, stream>>>(wq, wqb, DM * DM);
    cvt_bf16_kernel<<<1024, 256, 0, stream>>>(wk, wkb, DM * DM);
    cvt_bf16_kernel<<<1024, 256, 0, stream>>>(wv, wvb, DM * DM);
    cvt_bf16_kernel<<<1024, 256, 0, stream>>>(wo, wob, DM * DM);

    // 2) Q/K/V projections: (4096x1024) = (4096x1024) @ (1024x1024)^T
    int gblocks = (MROWS / 128) * (DM / 128);      // 256 workgroups
    gemm_nt_kernel<0><<<gblocks, 256, 0, stream>>>(xb, wqb, nullptr, Qs, MROWS, DM, DM);
    gemm_nt_kernel<0><<<gblocks, 256, 0, stream>>>(xb, wkb, nullptr, Ks, MROWS, DM, DM);
    gemm_nt_kernel<0><<<gblocks, 256, 0, stream>>>(xb, wvb, nullptr, Vs, MROWS, DM, DM);

    // 3) transpose V into head-major [B,H,hd,S] (xb region is free now)
    transpose_v_kernel<<<dim3(32, 32), 256, 0, stream>>>(Vs, Vt);

    // 4) attention: one wave per (b, h, 16-query block) -> 4096 waves
    int ablocks = (BB * NH * (SEQ / 16)) * 32 / 256;  // 512 blocks
    attn_kernel<<<ablocks, 256, 0, stream>>>(Qs, Ks, Vt, Os);

    // 5) output projection, fp32 result straight to d_out
    gemm_nt_kernel<1><<<gblocks, 256, 0, stream>>>(Os, wob, out, nullptr, MROWS, DM, DM);
}
